// MarlinLinear_33243046871187
// MI455X (gfx1250) — matrix operand order repaired and checked by reading, not compiled, not run on hardware
//
#include <hip/hip_runtime.h>

// MarlinLinear fused dequant-GEMM for gfx1250 (MI455X).
// out[R, N] = x[R, K] @ ((Wq - 8) * scales[k/G, n]) + bias,  R = B*S = 4096,
// K = M = 4096, N = 16384, G = 128.  f16 WMMA 16x16x32, f32 accumulate.
// BK = 64: two K=32 WMMA sub-steps per barrier phase (16 WMMAs/wave/phase).
// Dequant is bit-exact in packed fp16:
//   q in [0,15] lives in byte 0 of its int32, so one v_perm_b32 packs two q's
//   into a dword; OR 0x64006400 makes {1024+q1, 1024+q0} (exact);
//   pk_add(-1032) gives {q1-8, q0-8} (exact); pk_mul by scale matches the
//   reference's single fp16 rounding.

typedef __attribute__((ext_vector_type(16))) _Float16 v16h;
typedef __attribute__((ext_vector_type(8)))  _Float16 v8h;
typedef __attribute__((ext_vector_type(4)))  _Float16 v4h;
typedef __attribute__((ext_vector_type(2)))  _Float16 v2h;
typedef __attribute__((ext_vector_type(8)))  float    v8f;
typedef __attribute__((ext_vector_type(4)))  int      v4i;

#define RDIM 4096            // B*S output rows
#define KDIM 4096            // reduction dim (M in reference)
#define NDIM 16384           // output cols
#define GRP  128             // quant group size along K

#define BM 128               // block tile rows
#define BN 128               // block tile cols
#define BK 64                // K per barrier phase (2 x k=32 WMMA sub-steps)
#define SA 72                // LDS stride (halves) for A rows (padded, 144 B)
#define SB 72                // LDS stride (halves) for B columns (padded)

__launch_bounds__(256)
__global__ void marlin_wmma_kernel(const _Float16* __restrict__ x,
                                   const int*      __restrict__ Wq,
                                   const _Float16* __restrict__ scales,
                                   const _Float16* __restrict__ bias,
                                   float*          __restrict__ out)
{
    __shared__ _Float16 ldsA[2][BM * SA];   // row-major [row][k]      (36 KB)
    __shared__ _Float16 ldsB[2][BN * SB];   // col-major [n][k]        (36 KB)

    const int tid   = threadIdx.x;
    const int lane  = tid & 31;
    const int wave  = tid >> 5;         // 0..7
    const int wm    = wave >> 2;        // 0..1  (wave row)
    const int wn    = wave & 3;         // 0..3  (wave col)
    const int l15   = lane & 15;
    const int lhalf = lane >> 4;        // 0/1

    const int rowBlock = blockIdx.y * BM;
    const int colBlock = blockIdx.x * BN;

    // ---- global-load assignments (per thread, per K-tile) ----
    // A: thread t loads 32 halves: row = t/2, k offset = (t&1)*32
    const int ar = tid >> 1;
    const int ak = (tid & 1) * 32;
    const _Float16* xptr = x + (size_t)(rowBlock + ar) * KDIM + ak;

    // B: thread t covers 8 consecutive n and 4 consecutive k rows:
    //    n0 = (t&15)*8, kp = (t>>4)*4   -> 16*8 = 128 n,  16*4 = 64 k covered.
    const int nB = (tid & 15) * 8;
    const int kp = (tid >> 4) * 4;
    const int*      wqp   = Wq + (size_t)kp * NDIM + colBlock + nB;   // row kp
    const _Float16* scptr = scales + colBlock + nB;

    // staging registers
    v8h aR[4];
    v4i qR[4][2];       // [k row r][n 0..3 / n 4..7]
    v8h sR;             // 8 scales (one per n)

    auto fetch = [&](int k0) {
        const _Float16* ap = xptr + k0;
        aR[0] = *(const v8h*)(ap);
        aR[1] = *(const v8h*)(ap + 8);
        aR[2] = *(const v8h*)(ap + 16);
        aR[3] = *(const v8h*)(ap + 24);
#pragma unroll
        for (int r = 0; r < 4; ++r) {
            const int* p = wqp + (size_t)(k0 + r) * NDIM;
            qR[r][0] = *(const v4i*)(p);
            qR[r][1] = *(const v4i*)(p + 4);
        }
        // group index constant over the BK tile (BK divides GRP, k0 aligned)
        sR = *(const v8h*)(scptr + (size_t)(k0 >> 7) * NDIM);
    };

    auto stage = [&](int buf) {
        // A tile straight copy (16B-aligned b128 stores)
        *(v8h*)&ldsA[buf][ar * SA + ak]      = aR[0];
        *(v8h*)&ldsA[buf][ar * SA + ak + 8]  = aR[1];
        *(v8h*)&ldsA[buf][ar * SA + ak + 16] = aR[2];
        *(v8h*)&ldsA[buf][ar * SA + ak + 24] = aR[3];
        // B tile: packed dequant, 4 k per n -> one 8B ds_store_b64 each
        const v2h cM1032 = {(_Float16)-1032.0f, (_Float16)-1032.0f};
#pragma unroll
        for (int j = 0; j < 8; ++j) {
            const int sel = j >> 2, e = j & 3;
            const unsigned q0 = (unsigned)qR[0][sel][e];
            const unsigned q1 = (unsigned)qR[1][sel][e];
            const unsigned q2 = (unsigned)qR[2][sel][e];
            const unsigned q3 = (unsigned)qR[3][sel][e];
            const unsigned plo = __builtin_amdgcn_perm(q1, q0, 0x0C040C00u) | 0x64006400u;
            const unsigned phi = __builtin_amdgcn_perm(q3, q2, 0x0C040C00u) | 0x64006400u;
            const _Float16 s = sR[j];
            const v2h s2 = {s, s};
            const v2h w0 = (__builtin_bit_cast(v2h, plo) + cM1032) * s2; // k=kp,kp+1
            const v2h w1 = (__builtin_bit_cast(v2h, phi) + cM1032) * s2; // k=kp+2,kp+3
            *(v4h*)&ldsB[buf][(nB + j) * SB + kp] =
                __builtin_shufflevector(w0, w1, 0, 1, 2, 3);             // b64 store
        }
    };

    v8f acc[4][2];
#pragma unroll
    for (int i = 0; i < 4; ++i)
#pragma unroll
        for (int j = 0; j < 2; ++j)
            acc[i][j] = (v8f){0.f, 0.f, 0.f, 0.f, 0.f, 0.f, 0.f, 0.f};

    auto compute = [&](int buf) {
#pragma unroll
        for (int ks = 0; ks < 2; ++ks) {        // two k=32 sub-steps
            v16h bfrag[2];
            v16h afrag[4];
#pragma unroll
            for (int fn = 0; fn < 2; ++fn) {
                // lane = column (l15); lanes 0-15: K 0-7 & 16-23, lanes 16-31: K 8-15 & 24-31
                const _Float16* p =
                    &ldsB[buf][(wn * 32 + fn * 16 + l15) * SB + ks * 32 + lhalf * 8];
                v8h lo = *(const v8h*)(p);
                v8h hi = *(const v8h*)(p + 16);
                bfrag[fn] = __builtin_shufflevector(lo, hi,
                    0,1,2,3,4,5,6,7,8,9,10,11,12,13,14,15);
            }
#pragma unroll
            for (int fm = 0; fm < 4; ++fm) {
                // lane = row (l15); lanes 0-15: K 0-7 & 16-23, lanes 16-31: K 8-15 & 24-31
                const _Float16* p =
                    &ldsA[buf][(wm * 64 + fm * 16 + l15) * SA + ks * 32 + lhalf * 8];
                v8h lo = *(const v8h*)(p);
                v8h hi = *(const v8h*)(p + 16);
                afrag[fm] = __builtin_shufflevector(lo, hi,
                    0,1,2,3,4,5,6,7,8,9,10,11,12,13,14,15);
            }
#pragma unroll
            for (int fm = 0; fm < 4; ++fm)
#pragma unroll
                for (int fn = 0; fn < 2; ++fn)
                    acc[fm][fn] = __builtin_amdgcn_wmma_f32_16x16x32_f16(
                        /*neg_a=*/false, afrag[fm],
                        /*neg_b=*/false, bfrag[fn],
                        /*c_mod=*/(short)0, acc[fm][fn],
                        /*reuse_a=*/false, /*reuse_b=*/false);
        }
    };

    // ---- software-pipelined main loop (ping-pong LDS buffers) ----
    fetch(0);
    stage(0);
    const int nIters = KDIM / BK;   // 64
    for (int t = 0; t < nIters; ++t) {
        __syncthreads();            // buf[t&1] visible; prior readers of buf[(t+1)&1] done
        if (t + 1 < nIters) fetch((t + 1) * BK);
        compute(t & 1);
        if (t + 1 < nIters) stage((t + 1) & 1);
    }

    // ---- epilogue: bias add, f32 store ----
#pragma unroll
    for (int fn = 0; fn < 2; ++fn) {
        const int n = colBlock + wn * 32 + fn * 16 + l15;
        const float b = (float)bias[n];
#pragma unroll
        for (int fm = 0; fm < 4; ++fm) {
            const int rowb = rowBlock + wm * 64 + fm * 16 + 8 * lhalf;
            float* op = out + (size_t)rowb * NDIM + n;
#pragma unroll
            for (int i = 0; i < 8; ++i)       // C layout: VGPR i -> row (8*lhalf + i)
                op[(size_t)i * NDIM] = acc[fm][fn][i] + b;
        }
    }
}

extern "C" void kernel_launch(void* const* d_in, const int* in_sizes, int n_in,
                              void* d_out, int out_size, void* d_ws, size_t ws_size,
                              hipStream_t stream) {
    const _Float16* x      = (const _Float16*)d_in[0];
    const int*      Wq     = (const int*)d_in[1];
    const _Float16* scales = (const _Float16*)d_in[2];
    const _Float16* bias   = (const _Float16*)d_in[3];
    float*          out    = (float*)d_out;

    dim3 grid(NDIM / BN, RDIM / BM);   // 128 x 32 workgroups
    marlin_wmma_kernel<<<grid, 256, 0, stream>>>(x, Wq, scales, bias, out);
}
